// DMSRStyleConv_72782515798179
// MI455X (gfx1250) — matrix operand order repaired and checked by reading, not compiled, not run on hardware
//
#include <hip/hip_runtime.h>
#include <hip/hip_bf16.h>

typedef __attribute__((ext_vector_type(16))) _Float16 v16h;
typedef __attribute__((ext_vector_type(8)))  _Float16 v8h;
typedef __attribute__((ext_vector_type(4)))  _Float16 v4h;
typedef __attribute__((ext_vector_type(8)))  float    v8f;

#define BB    8
#define CIN   64
#define COUT  64
#define KS    3
#define SDIM  512
#define G     48
#define GO    46            // G - KS + 1
#define TAPS  27            // 3*3*3
#define KTOT  (CIN*TAPS)    // 1728
#define WPAD  72            // padded cin stride in LDS slab (16B-aligned rows)
#define DBLK  4             // d-outputs per block
#define PLANES (DBLK + KS - 1)           // 6 staged d-planes
#define NDB   ((GO + DBLK - 1) / DBLK)   // 12 d-blocks
#define SLAB_ELEMS (PLANES * KS * G * WPAD)   // 62208 f16
#define SLAB_BYTES (SLAB_ELEMS * 2)           // 124416 B (dynamic LDS, < 320KB/WGP)
#define PLANE_STRIDE (KS * G * WPAD)          // elements per d-plane group

union Frag { v16h f; v8h h[2]; };

#define WMMA_F16(A, B_, C) \
    __builtin_amdgcn_wmma_f32_16x16x32_f16(false, (A), false, (B_), (short)0, (C), false, false)

// ---------------------------------------------------------------------------
// Kernel 1: style modulation scales  s[b][cin] = style[b,:]·style_w[cin,:] + style_b[cin]
// ---------------------------------------------------------------------------
__global__ __launch_bounds__(256)
void style_kernel(const float* __restrict__ style, const float* __restrict__ sw,
                  const float* __restrict__ sb, float* __restrict__ s_out) {
    int i = blockIdx.x * blockDim.x + threadIdx.x;   // 0..511
    if (i >= BB * CIN) return;
    int b = i >> 6, cin = i & 63;
    float acc = sb[cin];
    const float* st = style + (size_t)b * SDIM;
    const float* wr = sw + (size_t)cin * SDIM;
    for (int j = 0; j < SDIM; ++j) acc += st[j] * wr[j];
    s_out[i] = acc;
}

// ---------------------------------------------------------------------------
// Kernel 2: modulate + demodulate weights, emit f16 in K-order k = tap*64 + cin
//           wmod[b][cout][tap][cin]
// ---------------------------------------------------------------------------
__global__ __launch_bounds__(128)
void modw_kernel(const float* __restrict__ weight, const float* __restrict__ s,
                 _Float16* __restrict__ wmod) {
    const int cout = blockIdx.x, b = blockIdx.y, t = threadIdx.x;
    __shared__ float s_sh[CIN];
    __shared__ float wtmp[KTOT];
    __shared__ float red[128];

    if (t < CIN) s_sh[t] = s[b * CIN + t];
    __syncthreads();

    const float* wrow = weight + (size_t)cout * KTOT;   // [cin][tap] contiguous
    float ssq = 0.f;
    for (int e = t; e < KTOT; e += 128) {
        int cin = e / TAPS;
        float v = wrow[e] * s_sh[cin];
        wtmp[e] = v;
        ssq += v * v;
    }
    red[t] = ssq;
    __syncthreads();
    for (int off = 64; off > 0; off >>= 1) {
        if (t < off) red[t] += red[t + off];
        __syncthreads();
    }
    const float r = rsqrtf(red[0] + 1e-8f);

    _Float16* wout = wmod + (size_t)(b * COUT + cout) * KTOT;
    for (int e = t; e < KTOT; e += 128) {
        int cin = e / TAPS, tap = e % TAPS;
        wout[tap * CIN + cin] = (_Float16)(wtmp[e] * r);
    }
}

// ---------------------------------------------------------------------------
// Kernel 3: implicit-GEMM conv, 2x2 register tiling per wave, FULLY UNROLLED
// K loop (27 taps x 2 cin-halves). Block = (b, 4 d-outputs, h row); 12 waves;
// wave owns 2 Cout-tiles x 2 d's -> 2 A frags + 2 B frags feed 4 WMMAs.
// ---------------------------------------------------------------------------
__global__ __launch_bounds__(384)
void conv_kernel(const float* __restrict__ x, const _Float16* __restrict__ wmod,
                 const float* __restrict__ bias, float* __restrict__ out) {
    extern __shared__ _Float16 slab[];   // [PLANES=6][KS=3][G=48][WPAD=72]

    const int b  = blockIdx.y;
    const int hh = blockIdx.x % GO;
    const int d0 = (blockIdx.x / GO) * DBLK;
    const int t  = threadIdx.x;

    // ---- stage x[b][cin][d0+dz (clamped)][hh+dy][0..47] -> slab[dz][dy][w][cin]
    {
        const float* xb = x + (size_t)b * CIN * G * G * G;
        const size_t cs = (size_t)G * G * G;
        for (int it = t; it < (CIN / 4) * PLANES * KS * G; it += 384) {  // 13824/384 = 36 exact
            int w    = it % G;
            int rest = it / G;
            int dy   = rest % KS;     rest /= KS;
            int dz   = rest % PLANES;
            int c4   = rest / PLANES;             // 0..15 -> cin base c4*4
            int dg   = d0 + dz; dg = dg > (G - 1) ? (G - 1) : dg;   // clamp dead tail planes
            const float* src = xb + (((size_t)(c4 * 4) * G + dg) * G + (hh + dy)) * G + w;
            v4h p;
            p[0] = (_Float16)src[0];
            p[1] = (_Float16)src[cs];
            p[2] = (_Float16)src[2 * cs];
            p[3] = (_Float16)src[3 * cs];
            *(v4h*)&slab[((dz * KS + dy) * G + w) * WPAD + c4 * 4] = p;
        }
    }
    __syncthreads();

    const int wave = t >> 5;
    const int lane = t & 31;
    const int l16  = lane & 15;
    const int hi   = lane >> 4;
    const int mp   = wave & 1;            // m0 = mp*32 -> Cout tiles m0, m0+16
    const int n_t  = (wave >> 1) % 3;     // W tile 0..2
    const int dp   = wave / 6;            // d pair 0/1 -> outputs d0+2*dp, +1
    const int m0   = mp * 32;
    const int w0   = n_t * 16;

    // A fragments: wmod[b][cout][tap][cin]; hi*8 folded into base.
    // Lanes 0-15 take K 0-7 & 16-23 of each 32-chunk, lanes 16-31 take 8-15 & 24-31.
    const _Float16* wb0 = wmod + (size_t)(b * COUT + m0 + l16) * KTOT + hi * 8;
    const _Float16* wb1 = wb0 + (size_t)16 * KTOT;

    // Precompute the 3 clamped column offsets (only runtime part of B addresses).
    int col[3];
#pragma unroll
    for (int kw = 0; kw < KS; ++kw) {
        int wl = w0 + l16 + kw;
        wl = wl > (G - 1) ? (G - 1) : wl;      // select; EXEC stays all-1s
        col[kw] = wl * WPAD + hi * 8;
    }
    const _Float16* slabp = slab + dp * 2 * PLANE_STRIDE;

    v8f acc00 = {}, acc01 = {}, acc10 = {}, acc11 = {};   // [m-tile][d-tile]
#pragma unroll
    for (int kd = 0; kd < KS; ++kd) {
#pragma unroll
        for (int kh = 0; kh < KS; ++kh) {
#pragma unroll
            for (int kw = 0; kw < KS; ++kw) {
                const int tap    = (kd * KS + kh) * KS + kw;
                const int rowoff = (kd * KS + kh) * G * WPAD;    // compile-time constant
                const _Float16* brow = slabp + col[kw];
                const _Float16* ar0  = wb0 + tap * CIN;
                const _Float16* ar1  = wb1 + tap * CIN;
#pragma unroll
                for (int ch = 0; ch < 2; ++ch) {
                    const int c0 = ch * 32;
                    Frag a0, a1, bA, bB;
                    a0.h[0] = *(const v8h*)(ar0 + c0);
                    a0.h[1] = *(const v8h*)(ar0 + c0 + 16);
                    a1.h[0] = *(const v8h*)(ar1 + c0);
                    a1.h[1] = *(const v8h*)(ar1 + c0 + 16);
                    bA.h[0] = *(const v8h*)(brow + rowoff + c0);
                    bA.h[1] = *(const v8h*)(brow + rowoff + c0 + 16);
                    bB.h[0] = *(const v8h*)(brow + rowoff + PLANE_STRIDE + c0);
                    bB.h[1] = *(const v8h*)(brow + rowoff + PLANE_STRIDE + c0 + 16);
                    acc00 = WMMA_F16(a0.f, bA.f, acc00);
                    acc10 = WMMA_F16(a1.f, bA.f, acc10);
                    acc01 = WMMA_F16(a0.f, bB.f, acc01);
                    acc11 = WMMA_F16(a1.f, bB.f, acc11);
                }
            }
        }
    }

    // ---- epilogue: bias add + guarded strided stores (no arrays -> no stack)
    // C/D layout: lane l -> N = l%16, row M = vgpr + 8*(l>=16)
    const int wout = w0 + l16;
    const size_t ostride = (size_t)GO * GO * GO;
    const int d_a = d0 + dp * 2;
    const int d_b = d_a + 1;
    const bool wok = wout < GO;
    const int cb0 = m0 + hi * 8;
    const int cb1 = m0 + 16 + hi * 8;
    const v8f bv0 = *(const v8f*)(bias + cb0);
    const v8f bv1 = *(const v8f*)(bias + cb1);

#define EMIT_TILE(ACC, BV, CB, DOUT)                                                  \
    do {                                                                              \
        if (wok && (DOUT) < GO) {                                                     \
            v8f r = (ACC) + (BV);                                                     \
            float* ob = out + ((((size_t)b * COUT + (CB)) * GO + (DOUT)) * GO + hh)   \
                              * GO + wout;                                            \
            for (int v = 0; v < 8; ++v) ob[(size_t)v * ostride] = r[v];               \
        }                                                                             \
    } while (0)

    EMIT_TILE(acc00, bv0, cb0, d_a);
    EMIT_TILE(acc01, bv0, cb0, d_b);
    EMIT_TILE(acc10, bv1, cb1, d_a);
    EMIT_TILE(acc11, bv1, cb1, d_b);
#undef EMIT_TILE
}

// ---------------------------------------------------------------------------
extern "C" void kernel_launch(void* const* d_in, const int* in_sizes, int n_in,
                              void* d_out, int out_size, void* d_ws, size_t ws_size,
                              hipStream_t stream) {
    const float* x      = (const float*)d_in[0];
    const float* style  = (const float*)d_in[1];
    const float* weight = (const float*)d_in[2];
    const float* bias   = (const float*)d_in[3];
    const float* sw     = (const float*)d_in[4];
    const float* sb     = (const float*)d_in[5];
    float* out = (float*)d_out;

    _Float16* wmod = (_Float16*)d_ws;                                   // 8*64*1728 f16 = 1.69 MB
    float*    s    = (float*)((char*)d_ws + (size_t)BB * COUT * KTOT * sizeof(_Float16));

    // Allow 124.4 KB dynamic LDS (CDNA5 WGP has 320 KB) — host-side config, capture-safe.
    (void)hipFuncSetAttribute(reinterpret_cast<const void*>(conv_kernel),
                              hipFuncAttributeMaxDynamicSharedMemorySize, SLAB_BYTES);

    style_kernel<<<2, 256, 0, stream>>>(style, sw, sb, s);
    modw_kernel<<<dim3(COUT, BB), 128, 0, stream>>>(weight, s, wmod);
    conv_kernel<<<dim3(GO * NDB, BB), 384, SLAB_BYTES, stream>>>(x, wmod, bias, out);
}
